// GRU_83133386982146
// MI455X (gfx1250) — matrix operand order repaired and checked by reading, not compiled, not run on hardware
//
#include <hip/hip_runtime.h>

// ---------------------------------------------------------------------------
// 2-layer GRU (H=10, B=1024, T=2048, in=1) + Linear(10->1), batch_first.
// Latency-bound sequential scan: one wave owns 16 batch rows, both layers
// fused, all state in registers. Recurrent matmuls computed transposed:
//   gates^T = W * h^T  with A = weights (constant regs), B = [h1 ; h2]^T.
//
// Combined B operand (f16 32x16, batch n = lane%16):
//   K = 0..4   -> h1[j=0..4]     K = 5..9   -> h2[j=0..4]    (lanes 0-15)
//   K = 16..20 -> h1[j=5..9]     K = 21..25 -> h2[j=5..9]    (lanes 16-31)
//   As v16h elements (lane-local): e=0..4 -> h1, e=5..9 -> h2, rest zero.
//
// Per step, 5 WMMAs:
//  pass1 (B = h1_{t-1}, h2_{t-1}):  P0,P1,P2 = { hr0, hz0, hn0, hn2 } rows
//  pass2 (B = h1_t,     h2_{t-1}):  Q0,Q1    = { r1(fused x+h), z1(fused), xn1 }
// Gate rows are permuted so that for each j, all its gate values land in the
// same lane half as h[j]: every elementwise op is lane-local. Biases ride in
// the WMMA C operand. Output projection needs one cross-half reduce per step
// (v_permlanex16, VALU — no DS counter).
// ---------------------------------------------------------------------------

typedef __attribute__((ext_vector_type(16))) _Float16 v16h;
typedef __attribute__((ext_vector_type(8)))  float    v8f;
typedef __attribute__((ext_vector_type(4)))  float    f32x4;

#define GRU_H 10
#define GRU_B 1024
#define GRU_T 2048

enum { TY_PAD = -1, TY_R0H = 0, TY_Z0H, TY_N0H, TY_N2H, TY_RS, TY_ZS, TY_XN };

// slot (pass, tile, g) -> row type + pair index i (i=0..4; half0 j=i, half1 j=5+i)
__device__ __forceinline__ void slot_map(int pass, int tile, int g, int& type, int& i) {
    type = TY_PAD; i = 0;
    if (pass == 1) {
        if (tile == 0)      { if (g <= 4) { type = TY_R0H; i = g; }     else { type = TY_Z0H; i = g - 5; } }
        else if (tile == 1) { if (g <= 1) { type = TY_Z0H; i = 3 + g; } else if (g <= 6) { type = TY_N0H; i = g - 2; } }
        else                { if (g <= 4) { type = TY_N2H; i = g; } }
    } else {
        if (tile == 0)      { if (g <= 4) { type = TY_RS; i = g; }      else { type = TY_ZS; i = g - 5; } }
        else                { if (g <= 1) { type = TY_ZS; i = 3 + g; }  else if (g <= 6) { type = TY_XN; i = g - 2; } }
    }
}

// B-operand K column -> which state (0=h1, 1=h2, -1=pad) and h index j
__device__ __forceinline__ int k_state(int K) {
    if (K <= 4) return 0;
    if (K <= 9) return 1;
    if (K >= 16 && K <= 20) return 0;
    if (K >= 21 && K <= 25) return 1;
    return -1;
}
__device__ __forceinline__ int k_col(int K) {
    if (K <= 4) return K;
    if (K <= 9) return K - 5;
    if (K >= 16 && K <= 20) return K - 11;  // j = 5..9
    return K - 16;                          // K=21..25 -> j = 5..9
}

__device__ __forceinline__ float wval(int type, int jrow, int state, int jcol,
                                      const float* __restrict__ Whh0,
                                      const float* __restrict__ Wih1,
                                      const float* __restrict__ Whh1) {
    switch (type) {
        case TY_R0H: return state == 0 ? Whh0[jrow * GRU_H + jcol] : 0.0f;
        case TY_Z0H: return state == 0 ? Whh0[(10 + jrow) * GRU_H + jcol] : 0.0f;
        case TY_N0H: return state == 0 ? Whh0[(20 + jrow) * GRU_H + jcol] : 0.0f;
        case TY_N2H: return state == 1 ? Whh1[(20 + jrow) * GRU_H + jcol] : 0.0f;
        case TY_RS:  return state == 0 ? Wih1[jrow * GRU_H + jcol]
                                       : Whh1[jrow * GRU_H + jcol];
        case TY_ZS:  return state == 0 ? Wih1[(10 + jrow) * GRU_H + jcol]
                                       : Whh1[(10 + jrow) * GRU_H + jcol];
        case TY_XN:  return state == 0 ? Wih1[(20 + jrow) * GRU_H + jcol] : 0.0f;
        default:     return 0.0f;
    }
}

__device__ __forceinline__ float cval(int type, int jrow,
                                      const float* __restrict__ bih0, const float* __restrict__ bhh0,
                                      const float* __restrict__ bih1, const float* __restrict__ bhh1) {
    switch (type) {
        case TY_R0H: return bih0[jrow] + bhh0[jrow];
        case TY_Z0H: return bih0[10 + jrow] + bhh0[10 + jrow];
        case TY_N0H: return bhh0[20 + jrow];
        case TY_N2H: return bhh1[20 + jrow];
        case TY_RS:  return bih1[jrow] + bhh1[jrow];
        case TY_ZS:  return bih1[10 + jrow] + bhh1[10 + jrow];
        case TY_XN:  return bih1[20 + jrow];
        default:     return 0.0f;
    }
}

// A tile build (16x32 f16). Layout: lane L (row M = L%16):
//   lanes 0-15 : K=e     (0..15)
//   lanes 16-31: K=e+16  (16..31)
__device__ __forceinline__ v16h build_A(int pass, int tile, int lane,
                                        const float* __restrict__ Whh0,
                                        const float* __restrict__ Wih1,
                                        const float* __restrict__ Whh1) {
    v16h a;
    const int  M    = lane & 15;
    const int  g    = M & 7;
    const int  half = M >> 3;
    const bool hiK  = lane >= 16;
    int type, i;
    slot_map(pass, tile, g, type, i);
    const int jrow = half ? 5 + i : i;
#pragma unroll
    for (int e = 0; e < 16; ++e) {
        int K  = hiK ? e + 16 : e;
        int st = k_state(K);
        float w = 0.0f;
        if (type != TY_PAD && st >= 0)
            w = wval(type, jrow, st, k_col(K), Whh0, Wih1, Whh1);
        a[e] = (_Float16)w;
    }
    return a;
}

__device__ __forceinline__ v8f build_C(int pass, int tile, int lane,
                                       const float* __restrict__ bih0, const float* __restrict__ bhh0,
                                       const float* __restrict__ bih1, const float* __restrict__ bhh1) {
    v8f c;
    const int half = lane >> 4;
#pragma unroll
    for (int g = 0; g < 8; ++g) {
        int type, i;
        slot_map(pass, tile, g, type, i);
        int jrow = half ? 5 + i : i;
        c[g] = (type == TY_PAD) ? 0.0f : cval(type, jrow, bih0, bhh0, bih1, bhh1);
    }
    return c;
}

#if __has_builtin(__builtin_amdgcn_tanhf)
__device__ __forceinline__ float tanh_t(float x) { return __builtin_amdgcn_tanhf(x); }
#else
__device__ __forceinline__ float tanh_t(float x) {
    float e = __builtin_amdgcn_exp2f(x * -2.88539008f);
    return __builtin_amdgcn_rcpf(1.0f + e) * 2.0f - 1.0f;
}
#endif
__device__ __forceinline__ float sigm(float x) {
    return fmaf(0.5f, tanh_t(0.5f * x), 0.5f);
}

__device__ __forceinline__ float xhalf_sum(float v) {
#if __has_builtin(__builtin_amdgcn_permlanex16)
    int o = __builtin_amdgcn_permlanex16(__float_as_int(v), __float_as_int(v),
                                         0x76543210, 0xFEDCBA98, false, false);
    return v + __int_as_float(o);
#else
    int o = __builtin_amdgcn_ds_swizzle(__float_as_int(v), 0x401F); // SWAPX16
    return v + __int_as_float(o);
#endif
}

__global__ __launch_bounds__(32)
void gru2_wmma_kernel(const float* __restrict__ x,
                      const float* __restrict__ Wih0, const float* __restrict__ Whh0,
                      const float* __restrict__ bih0, const float* __restrict__ bhh0,
                      const float* __restrict__ Wih1, const float* __restrict__ Whh1,
                      const float* __restrict__ bih1, const float* __restrict__ bhh1,
                      const float* __restrict__ Wlin, const float* __restrict__ blin,
                      float* __restrict__ out) {
    const int lane  = threadIdx.x;           // one wave per block
    const int hi    = lane >> 4;
    const int bglob = blockIdx.x * 16 + (lane & 15);

    // ---- constant A tiles ----
    const v16h A10 = build_A(1, 0, lane, Whh0, Wih1, Whh1);
    const v16h A11 = build_A(1, 1, lane, Whh0, Wih1, Whh1);
    const v16h A12 = build_A(1, 2, lane, Whh0, Wih1, Whh1);
    const v16h A20 = build_A(2, 0, lane, Whh0, Wih1, Whh1);
    const v16h A21 = build_A(2, 1, lane, Whh0, Wih1, Whh1);
    // ---- constant C tiles (biases) ----
    const v8f C10 = build_C(1, 0, lane, bih0, bhh0, bih1, bhh1);
    const v8f C11 = build_C(1, 1, lane, bih0, bhh0, bih1, bhh1);
    const v8f C12 = build_C(1, 2, lane, bih0, bhh0, bih1, bhh1);
    const v8f C20 = build_C(2, 0, lane, bih0, bhh0, bih1, bhh1);
    const v8f C21 = build_C(2, 1, lane, bih0, bhh0, bih1, bhh1);

    // ---- layer-0 x-side + output-head per-slot constants ----
    float wr0[5], wz0[5], wn0[5], bn0[5], wl[5];
#pragma unroll
    for (int i = 0; i < 5; ++i) {
        int j  = hi ? 5 + i : i;
        wr0[i] = Wih0[j];
        wz0[i] = Wih0[10 + j];
        wn0[i] = Wih0[20 + j];
        bn0[i] = bih0[20 + j];
        wl[i]  = Wlin[j];
    }
    const float blin_v = blin[0];

    // ---- state ----
    float h1v[5], h2v[5];
    v16h  Bc;
#pragma unroll
    for (int e = 0; e < 16; ++e) Bc[e] = (_Float16)0.0f;
#pragma unroll
    for (int i = 0; i < 5; ++i) { h1v[i] = 0.0f; h2v[i] = 0.0f; }

    const f32x4* __restrict__ x4 = (const f32x4*)(x + (size_t)bglob * GRU_T);
    f32x4*       __restrict__ o4 = (f32x4*)(out + (size_t)bglob * GRU_T);

    const int N4 = GRU_T / 4;
    f32x4 xq = x4[0];                          // prologue load
    for (int t4 = 0; t4 < N4; ++t4) {
        // software pipeline: issue next iteration's load now, wait at the bottom
        const int tn = (t4 + 1 < N4) ? t4 + 1 : t4;
        f32x4 xnext = x4[tn];
        __builtin_prefetch((const void*)(x4 + ((t4 + 4 < N4) ? t4 + 4 : t4)), 0, 1);

        f32x4 ov;
#pragma unroll
        for (int s = 0; s < 4; ++s) {
            const float xt = xq[s];

            // ---- pass 1: uses (h1_{t-1}, h2_{t-1}) ----
            v8f P0 = __builtin_amdgcn_wmma_f32_16x16x32_f16(false, A10, false, Bc, (short)0, C10, false, false);
            v8f P1 = __builtin_amdgcn_wmma_f32_16x16x32_f16(false, A11, false, Bc, (short)0, C11, false, false);
            v8f P2 = __builtin_amdgcn_wmma_f32_16x16x32_f16(false, A12, false, Bc, (short)0, C12, false, false);

            // ---- layer 0 elementwise ----
#pragma unroll
            for (int i = 0; i < 5; ++i) {
                float r  = sigm(fmaf(xt, wr0[i], P0[i]));
                float zp = (i < 3) ? P0[5 + i] : P1[i - 3];
                float z  = sigm(fmaf(xt, wz0[i], zp));
                float n  = tanh_t(fmaf(r, P1[2 + i], fmaf(xt, wn0[i], bn0[i])));
                h1v[i]   = n + z * (h1v[i] - n);
            }
#pragma unroll
            for (int i = 0; i < 5; ++i) Bc[i] = (_Float16)h1v[i];

            // ---- pass 2: uses (h1_t, h2_{t-1}); r/z arrive pre-summed ----
            v8f Q0 = __builtin_amdgcn_wmma_f32_16x16x32_f16(false, A20, false, Bc, (short)0, C20, false, false);
            v8f Q1 = __builtin_amdgcn_wmma_f32_16x16x32_f16(false, A21, false, Bc, (short)0, C21, false, false);

            // ---- layer 1 elementwise + output head ----
            float acc = 0.0f;
#pragma unroll
            for (int i = 0; i < 5; ++i) {
                float r = sigm(Q0[i]);
                float z = sigm((i < 3) ? Q0[5 + i] : Q1[i - 3]);
                float n = tanh_t(fmaf(r, P2[i], Q1[2 + i]));
                float h = n + z * (h2v[i] - n);
                h2v[i]  = h;
                Bc[5 + i] = (_Float16)h;
                acc = fmaf(wl[i], h, acc);
            }
            ov[s] = xhalf_sum(acc) + blin_v;   // half0 j=0..4 + half1 j=5..9
        }
        if (lane < 16) o4[t4] = ov;
        xq = xnext;
    }
}

extern "C" void kernel_launch(void* const* d_in, const int* in_sizes, int n_in,
                              void* d_out, int out_size, void* d_ws, size_t ws_size,
                              hipStream_t stream) {
    (void)in_sizes; (void)n_in; (void)out_size; (void)d_ws; (void)ws_size;
    const float* x    = (const float*)d_in[0];
    const float* Wih0 = (const float*)d_in[1];
    const float* Whh0 = (const float*)d_in[2];
    const float* bih0 = (const float*)d_in[3];
    const float* bhh0 = (const float*)d_in[4];
    const float* Wih1 = (const float*)d_in[5];
    const float* Whh1 = (const float*)d_in[6];
    const float* bih1 = (const float*)d_in[7];
    const float* bhh1 = (const float*)d_in[8];
    const float* Wlin = (const float*)d_in[9];
    const float* blin = (const float*)d_in[10];

    dim3 grid(GRU_B / 16);   // 64 blocks, one wave each (16 batch rows / wave)
    dim3 block(32);
    gru2_wmma_kernel<<<grid, block, 0, stream>>>(
        x, Wih0, Whh0, bih0, bhh0, Wih1, Whh1, bih1, bhh1, Wlin, blin,
        (float*)d_out);
}
